// BahdanauAttention_60095182406037
// MI455X (gfx1250) — compile-verified, matrix operand order checked by reading, not run on hardware
//
#include <hip/hip_runtime.h>
#include <cmath>

#define B_ 32
#define S_ 4096
#define H_ 512
#define E_ 512
#define NEGV (-1000000000.0f)
#define CHUNK 256
#define NCHUNK (S_ / CHUNK) /* 16 */
#define KS 520              /* padded bf16 stride: 1040B -> rows offset by 4 banks */

typedef __attribute__((ext_vector_type(16))) __bf16 v16bf;
typedef __attribute__((ext_vector_type(8)))  __bf16 v8bf;
typedef __attribute__((ext_vector_type(8)))  float  v8f;

// Native v_tanh_f32 if the toolchain exposes it (CDNA5 lists TANH as a TRANS op);
// fall back to libm expansion otherwise so the compile can never break.
#if __has_builtin(__builtin_amdgcn_tanhf)
#define FAST_TANH(x) __builtin_amdgcn_tanhf(x)
#elif __has_builtin(__builtin_amdgcn_tanh_f32)
#define FAST_TANH(x) __builtin_amdgcn_tanh_f32(x)
#else
#define FAST_TANH(x) tanhf(x)
#endif

// ---------------- Kernel 1: qpk[b][h] = query[b]@Wq + bq + bk (fold k-proj bias in) ---
__global__ __launch_bounds__(256, 1)
void qproj_kernel(const float* __restrict__ query, const float* __restrict__ Wq,
                  const float* __restrict__ bq, const float* __restrict__ bk,
                  float* __restrict__ qpk) {
  const int b = blockIdx.x;
  const int t = threadIdx.x;
  __shared__ float qLDS[H_];
  qLDS[t]       = query[b * H_ + t];
  qLDS[t + 256] = query[b * H_ + t + 256];
  __syncthreads();
  for (int h = t; h < H_; h += 256) {
    float acc = 0.f;
    for (int e = 0; e < H_; ++e) acc += qLDS[e] * Wq[e * H_ + h];
    qpk[b * H_ + h] = acc + bq[h] + bk[h];
  }
}

// ---------------- Kernel 2: fused scores (WMMA bf16 GEMM + tanh/v-dot) + chunk softmax
// ----------------           partials + partial context -------------------------------
__global__ __launch_bounds__(256, 1)
void score_ctx_kernel(const float* __restrict__ keys, const int* __restrict__ mask,
                      const float* __restrict__ Wk, const float* __restrict__ qpk,
                      const float* __restrict__ v, const float* __restrict__ bv,
                      float* __restrict__ Praw, float* __restrict__ Mpart,
                      float* __restrict__ Lpart, float* __restrict__ Cpart) {
  __shared__ __align__(16) __bf16 keysLDS[CHUNK * KS]; // ~260 KB
  __shared__ __align__(16) __bf16 wkT[16 * KS];        // Wk column-block, transposed
  __shared__ float qpkLDS[H_];
  __shared__ float vLDS[H_];
  __shared__ float scoresLDS[CHUNK];
  __shared__ float pLDS[CHUNK];

  const int b     = blockIdx.x / NCHUNK;
  const int chunk = blockIdx.x % NCHUNK;
  const int s0    = chunk * CHUNK;
  const int t     = threadIdx.x;
  const int wave  = t >> 5;
  const int lane  = t & 31;

  // ---- stage keys chunk -> LDS (fp32 global, coalesced float4; bf16 packed b64) ----
  const float* kbase = keys + ((size_t)b * S_ + s0) * (size_t)E_;
  for (int j = t * 4; j < CHUNK * E_; j += 256 * 4) {
    const int row = j / E_, col = j % E_;
    const float4 f = *reinterpret_cast<const float4*>(kbase + (size_t)row * E_ + col);
    union { __bf16 h[4]; uint2 u; } pk;
    pk.h[0] = (__bf16)f.x; pk.h[1] = (__bf16)f.y;
    pk.h[2] = (__bf16)f.z; pk.h[3] = (__bf16)f.w;
    *reinterpret_cast<uint2*>(&keysLDS[row * KS + col]) = pk.u; // 8B-aligned (KS even)
  }
  qpkLDS[t]       = qpk[b * H_ + t];
  qpkLDS[t + 256] = qpk[b * H_ + t + 256];
  vLDS[t]         = v[t];
  vLDS[t + 256]   = v[t + 256];
  __syncthreads();

  // each wave owns two 16-row M tiles of the 256-row chunk
  const int mts[2] = { wave, wave + 8 };
  // per-LANE partial of sum_h v[h]*tanh(qk[h]): each lane accumulates its N-slice
  // across all n-tiles; cross-lane reduction happens ONCE at the end.
  float scoreAcc[2][8];
#pragma unroll
  for (int mi = 0; mi < 2; ++mi)
#pragma unroll
    for (int r = 0; r < 8; ++r) scoreAcc[mi][r] = 0.f;

  const int kb_off = (lane & 16) ? 8 : 0; // 16-bit operand K split across lane halves
  const int nl     = lane & 15;           // N (for B/C) and M (for A) within tile

  const __bf16* arow0 = &keysLDS[(mts[0] * 16 + nl) * KS];
  const __bf16* arow1 = &keysLDS[(mts[1] * 16 + nl) * KS];

  for (int nt = 0; nt < H_ / 16; ++nt) {
    __syncthreads();
    // stage Wk[:, nt*16 .. nt*16+16) transposed: wkT[nn][k], k contiguous
    for (int j = t; j < 16 * E_; j += 256) {
      const int nn = j & 15, k = j >> 4;
      wkT[nn * KS + k] = (__bf16)Wk[k * H_ + nt * 16 + nn];
    }
    __syncthreads();

    v8f accs[2];
#pragma unroll
    for (int mi = 0; mi < 2; ++mi)
#pragma unroll
      for (int r = 0; r < 8; ++r) accs[mi][r] = 0.f;

    const __bf16* brow = &wkT[nl * KS];
#pragma unroll 4
    for (int k0 = 0; k0 < E_; k0 += 32) {
      const int kb = k0 + kb_off;
      const v8bf blo = *reinterpret_cast<const v8bf*>(brow + kb);
      const v8bf bhi = *reinterpret_cast<const v8bf*>(brow + kb + 16);
      const v16bf Bv = __builtin_shufflevector(blo, bhi,
          0, 1, 2, 3, 4, 5, 6, 7, 8, 9, 10, 11, 12, 13, 14, 15);

      const v8bf a0l = *reinterpret_cast<const v8bf*>(arow0 + kb);
      const v8bf a0h = *reinterpret_cast<const v8bf*>(arow0 + kb + 16);
      const v16bf A0 = __builtin_shufflevector(a0l, a0h,
          0, 1, 2, 3, 4, 5, 6, 7, 8, 9, 10, 11, 12, 13, 14, 15);
      accs[0] = __builtin_amdgcn_wmma_f32_16x16x32_bf16(
          false, A0, false, Bv, (short)0, accs[0], false, false);

      const v8bf a1l = *reinterpret_cast<const v8bf*>(arow1 + kb);
      const v8bf a1h = *reinterpret_cast<const v8bf*>(arow1 + kb + 16);
      const v16bf A1 = __builtin_shufflevector(a1l, a1h,
          0, 1, 2, 3, 4, 5, 6, 7, 8, 9, 10, 11, 12, 13, 14, 15);
      accs[1] = __builtin_amdgcn_wmma_f32_16x16x32_bf16(
          false, A1, false, Bv, (short)0, accs[1], false, false);
    }

    // epilogue: qk -> tanh -> *v[n]; accumulate per-lane (no cross-lane ops here)
    const float qv = qpkLDS[nt * 16 + nl];
    const float vv = vLDS[nt * 16 + nl];
#pragma unroll
    for (int mi = 0; mi < 2; ++mi)
#pragma unroll
      for (int r = 0; r < 8; ++r)
        scoreAcc[mi][r] += FAST_TANH(accs[mi][r] + qv) * vv;
  }

  // single cross-lane reduction over the 16 N-lanes of each half-wave
#pragma unroll
  for (int mi = 0; mi < 2; ++mi)
#pragma unroll
    for (int r = 0; r < 8; ++r) {
      float c = scoreAcc[mi][r];
      c += __shfl_xor(c, 1, 32);
      c += __shfl_xor(c, 2, 32);
      c += __shfl_xor(c, 4, 32);
      c += __shfl_xor(c, 8, 32);
      scoreAcc[mi][r] = c;
    }

  // write masked scores: lanes 0/16 hold row sums (M=r / M=r+8 of each tile)
  if ((lane & 15) == 0) {
    const float bvVal = bv[0];
    const int half = (lane >> 4) * 8;
#pragma unroll
    for (int mi = 0; mi < 2; ++mi) {
      const int base = mts[mi] * 16 + half;
#pragma unroll
      for (int r = 0; r < 8; ++r) {
        const int sl = base + r;
        float sc = scoreAcc[mi][r] + bvVal;
        if (mask[b * S_ + s0 + sl] == 0) sc = NEGV;
        scoresLDS[sl] = sc;
      }
    }
  }
  __syncthreads();

  // chunk-local softmax partials (broadcast LDS scans are cheap at 256 elems)
  float mx = -3.402823466e38f;
  for (int i = 0; i < CHUNK; ++i) mx = fmaxf(mx, scoresLDS[i]);
  const float p = __expf(scoresLDS[t] - mx);
  pLDS[t] = p;
  __syncthreads();
  float l = 0.f;
  for (int i = 0; i < CHUNK; ++i) l += pLDS[i];

  Praw[b * S_ + s0 + t] = p;
  if (t == 0) {
    Mpart[b * NCHUNK + chunk] = mx;
    Lpart[b * NCHUNK + chunk] = l;
  }

  // partial context: c[e] = sum_s p_s * keys[s][e] (fp32, coalesced, L2-hot)
  float c0 = 0.f, c1 = 0.f;
  for (int s = 0; s < CHUNK; ++s) {
    const float w = pLDS[s];
    const float* kr = kbase + (size_t)s * E_;
    c0 += w * kr[t];
    c1 += w * kr[t + 256];
  }
  float* cp = Cpart + ((size_t)(b * NCHUNK + chunk)) * E_;
  cp[t] = c0;
  cp[t + 256] = c1;
}

// ---------------- Kernel 3: combine chunk partials; emit attn + context --------------
__global__ __launch_bounds__(256, 1)
void finalize_kernel(const float* __restrict__ Praw, const float* __restrict__ Mpart,
                     const float* __restrict__ Lpart, const float* __restrict__ Cpart,
                     float* __restrict__ outCtx, float* __restrict__ outAttn) {
  const int b = blockIdx.x;
  const int t = threadIdx.x;
  __shared__ float sm[NCHUNK], sl[NCHUNK], scale[NCHUNK];
  __shared__ float gm_s, gl_s;
  if (t < NCHUNK) {
    sm[t] = Mpart[b * NCHUNK + t];
    sl[t] = Lpart[b * NCHUNK + t];
  }
  __syncthreads();
  if (t == 0) {
    float m = -3.402823466e38f;
    for (int c = 0; c < NCHUNK; ++c) m = fmaxf(m, sm[c]);
    float l = 0.f;
    for (int c = 0; c < NCHUNK; ++c) l += sl[c] * __expf(sm[c] - m);
    gm_s = m;
    gl_s = l;
  }
  __syncthreads();
  if (t < NCHUNK) scale[t] = __expf(sm[t] - gm_s) / gl_s;
  __syncthreads();
  for (int s = t; s < S_; s += 256)
    outAttn[b * S_ + s] = Praw[b * S_ + s] * scale[s / CHUNK];
  for (int e = t; e < E_; e += 256) {
    float a = 0.f;
    for (int c = 0; c < NCHUNK; ++c)
      a += Cpart[((size_t)(b * NCHUNK + c)) * E_ + e] * scale[c];
    outCtx[b * E_ + e] = a;
  }
}

extern "C" void kernel_launch(void* const* d_in, const int* in_sizes, int n_in,
                              void* d_out, int out_size, void* d_ws, size_t ws_size,
                              hipStream_t stream) {
  const float* query = (const float*)d_in[0];
  const float* keys  = (const float*)d_in[1];
  const int*   mask  = (const int*)d_in[2];
  const float* Wq    = (const float*)d_in[3];
  const float* bq    = (const float*)d_in[4];
  const float* Wk    = (const float*)d_in[5];
  const float* bk    = (const float*)d_in[6];
  const float* v     = (const float*)d_in[7];
  const float* bv    = (const float*)d_in[8];

  float* out     = (float*)d_out;
  float* outCtx  = out;            // B*E = 16384
  float* outAttn = out + B_ * E_;  // B*S = 131072

  float* ws   = (float*)d_ws;
  float* qpk  = ws;                        // B*H       = 16384
  float* Praw = qpk + B_ * H_;             // B*S       = 131072
  float* Mp   = Praw + B_ * S_;            // B*NCHUNK  = 512
  float* Lp   = Mp + B_ * NCHUNK;          // B*NCHUNK  = 512
  float* Cp   = Lp + B_ * NCHUNK;          // B*NCHUNK*E = 262144  (~1.6 MB total)

  qproj_kernel<<<B_, 256, 0, stream>>>(query, Wq, bq, bk, qpk);
  score_ctx_kernel<<<B_ * NCHUNK, 256, 0, stream>>>(keys, mask, Wk, qpk, v, bv,
                                                    Praw, Mp, Lp, Cp);
  finalize_kernel<<<B_, 256, 0, stream>>>(Praw, Mp, Lp, Cp, outCtx, outAttn);
}
